// HierarchicalRNNDecoder_66984309948813
// MI455X (gfx1250) — matrix operand order repaired and checked by reading, not compiled, not run on hardware
//
#include <hip/hip_runtime.h>
#include <hip/hip_bf16.h>

#define DEVINL __device__ __forceinline__

typedef __bf16 bf16_t;
typedef __bf16 v16bf __attribute__((ext_vector_type(16)));
typedef float  v8f   __attribute__((ext_vector_type(8)));

// ---------------- problem constants ----------------
constexpr int T_STEPS = 50;
constexpr int B       = 64;
constexpr int D       = 512;
constexpr int NE      = 64;    // N_ENTS
constexpr int ES      = 32;    // ENT_SIZE
constexpr int NU      = 2048;  // N_UNITS
constexpr int G4      = 2048;  // 4*D (gates)
constexpr int D2      = 1024;  // 2*D
constexpr float NEGINF = -1e9f;

// ---------------- launch constants ----------------
constexpr int NWG      = 64;
constexpr int TPB      = 256;                 // 8 wave32s
constexpr int NWAVES   = NWG * (TPB / 32);    // 512
constexpr int NTHREADS = NWG * TPB;           // 16384

// ---------------- CDNA5 async-to-LDS availability (guarded probe) --------
#if __has_builtin(__builtin_amdgcn_global_load_async_to_lds_b128) && \
    __has_builtin(__builtin_amdgcn_s_wait_asynccnt)
#define HAVE_ASYNC_LDS 1
#else
#define HAVE_ASYNC_LDS 0
#endif

DEVINL void async_copy_b128(const void* g, void* l) {
#if HAVE_ASYNC_LDS
  typedef int i32x4 __attribute__((vector_size(16)));
  typedef __attribute__((address_space(1))) i32x4* gptr_t;
  typedef __attribute__((address_space(3))) i32x4* lptr_t;
  __builtin_amdgcn_global_load_async_to_lds_b128((gptr_t)g, (lptr_t)l, 0, 0);
#else
  (void)g; (void)l;
#endif
}
DEVINL void wait_async0() {
#if HAVE_ASYNC_LDS
  __builtin_amdgcn_s_wait_asynccnt(0);
#endif
}

// ---------------- workspace pointer bundle ----------------
struct WS {
  bf16_t *embT;     // [T][B][D]     gathered+converted embeddings
  bf16_t *WihT0;    // packed 32x16 B-frag tiles, K=1024 N=2048
  bf16_t *WhhT0;    // K=512 N=2048
  bf16_t *WihT1;    // K=512 N=2048
  bf16_t *WhhT1;    // K=512 N=2048
  bf16_t *Wqu;      // K=512 N=512
  bf16_t *Wqc;      // K=512 N=512
  bf16_t *WoutT;    // K=1024 N=512
  bf16_t *h0bf;     // [B][D]
  bf16_t *h1bf;     // [B][D]   (== q in bf16)
  bf16_t *feedbf;   // [B][D]
  bf16_t *ctxqbf;   // [B][2D]  concat(ctx, q) bf16
  float  *g0;       // [B][4D]
  float  *g1;       // [B][4D]
  float  *c0;       // [B][D]
  float  *c1;       // [B][D]
  float  *qWu;      // [B][D]
  float  *qWc;      // [B][D]
  float  *su;       // [B][NU]
  float  *sc;       // [B][NE]
  float  *au;       // [B][NU]
  float  *ac;       // [B][NE]
  unsigned *cnt;    // grid barrier counter
};

// ---------------- helpers ----------------
DEVINL float sigmoidf_(float x) { return 1.0f / (1.0f + expf(-x)); }

// packed B-fragment address: tiles of 32(K)x16(N), per-lane 16 contiguous elems
DEVINL size_t packW(int k, int j, int ntiles) {
  size_t tile = (size_t)(k >> 5) * ntiles + (j >> 4);
  int    off  = (((k >> 3) & 1) * 16 + (j & 15)) * 16 + ((k >> 4) & 1) * 8 + (k & 7);
  return tile * 512 + off;
}

// A fragment (16x32 bf16, row-major source, ld in elements)
DEVINL v16bf load_a_frag(const bf16_t* a, int ld, int lane) {
  int row = lane & 15;
  int kb  = (lane < 16) ? 0 : 8;
  const float4* p = reinterpret_cast<const float4*>(a + (size_t)row * ld + kb);
  union { float4 f[2]; v16bf v; } u;
  u.f[0] = p[0];   // K kb..kb+7
  u.f[1] = p[2];   // K kb+16..kb+23
  return u.v;
}

// B fragment from pre-packed tile
DEVINL v16bf load_b_frag(const bf16_t* tile, int lane) {
  const float4* p = reinterpret_cast<const float4*>(tile + (size_t)lane * 16);
  union { float4 f[2]; v16bf v; } u;
  u.f[0] = p[0];
  u.f[1] = p[1];
  return u.v;
}

// C/D fragment store (16x16 f32)
DEVINL void store_c_frag(float* c, int ld, int lane, v8f acc) {
  int col = lane & 15;
  int rb  = (lane < 16) ? 0 : 8;
#pragma unroll
  for (int r = 0; r < 8; ++r) c[(size_t)(rb + r) * ld + col] = acc[r];
}

DEVINL v8f wmma_bf16(v16bf a, v16bf b, v8f c) {
  return __builtin_amdgcn_wmma_f32_16x16x32_bf16(false, a, false, b, (short)0, c,
                                                 false, false);
}

DEVINL float wave_sum(float v) {
#pragma unroll
  for (int m = 16; m >= 1; m >>= 1) v += __shfl_xor(v, m, 32);
  return v;
}
DEVINL float wave_max(float v) {
#pragma unroll
  for (int m = 16; m >= 1; m >>= 1) v = fmaxf(v, __shfl_xor(v, m, 32));
  return v;
}

// vectorized D=512 row dots (wave-cooperative)
DEVINL float rowdot_bf16(const float* qp, const bf16_t* kp, int lane) {
  float s = 0.0f;
#pragma unroll
  for (int it = 0; it < 2; ++it) {
    int c = it * 32 + lane;  // 16-byte chunk id, 64 chunks of 8 elems
    union { float4 f; bf16_t h[8]; } kc;
    kc.f = reinterpret_cast<const float4*>(kp)[c];
    union { float4 f[2]; float q[8]; } qc;
    const float4* qv = reinterpret_cast<const float4*>(qp + c * 8);
    qc.f[0] = qv[0];
    qc.f[1] = qv[1];
#pragma unroll
    for (int i = 0; i < 8; ++i) s = fmaf(qc.q[i], (float)kc.h[i], s);
  }
  return wave_sum(s);
}
DEVINL float rowdot_f32(const float* qp, const float* kp, int lane) {
  float s = 0.0f;
  for (int d = lane; d < D; d += 32) s = fmaf(qp[d], kp[d], s);
  return wave_sum(s);
}

// monotonic grid barrier (counter zeroed by prep kernel each launch)
DEVINL void grid_sync(unsigned* cnt, unsigned& gen) {
  __syncthreads();
  gen += NWG;
  if (threadIdx.x == 0) {
    __threadfence();
    atomicAdd(cnt, 1u);
    volatile unsigned* vc = (volatile unsigned*)cnt;
    while (*vc < gen) __builtin_amdgcn_s_sleep(1);
  }
  __syncthreads();
  __threadfence();
}

// ---------------- prep kernels ----------------
__global__ void prep_state_kernel(WS ws, const float* __restrict__ enc,
                                  const float* __restrict__ sif,
                                  float* __restrict__ outs) {
  int idx = blockIdx.x * blockDim.x + threadIdx.x;
  if (idx == 0) *ws.cnt = 0u;
  if (idx < B * D) {
    int d = idx & (D - 1);
    float e = enc[idx];
    ws.c0[idx] = e;
    ws.c1[idx] = e;
    bf16_t eb = (bf16_t)e;
    ws.h0bf[idx] = eb;
    ws.h1bf[idx] = eb;
    float f0 = sif[d];
    ws.feedbf[idx] = (bf16_t)f0;
    outs[idx] = f0;  // dec_outs[0]
  }
}

__global__ void prep_emb_kernel(WS ws, const int* __restrict__ tgt,
                                const float* __restrict__ emb_table) {
  const size_t stride = (size_t)gridDim.x * blockDim.x;
  for (size_t idx = (size_t)blockIdx.x * blockDim.x + threadIdx.x;
       idx < (size_t)T_STEPS * B * D; idx += stride) {
    size_t tb = idx >> 9;
    int d = (int)(idx & (D - 1));
    int tok = tgt[tb];
    ws.embT[idx] = (bf16_t)emb_table[(size_t)tok * D + d];
  }
}

__global__ void prep_weights_kernel(WS ws, const float* __restrict__ pWih0,
                                    const float* __restrict__ pWhh0,
                                    const float* __restrict__ pWih1,
                                    const float* __restrict__ pWhh1,
                                    const float* __restrict__ pWqu,
                                    const float* __restrict__ pWqc,
                                    const float* __restrict__ pWout) {
  const size_t stride = (size_t)gridDim.x * blockDim.x;
  const size_t t0 = (size_t)blockIdx.x * blockDim.x + threadIdx.x;

  for (size_t i = t0; i < (size_t)D2 * G4; i += stride) {
    int k = (int)(i / G4), j = (int)(i % G4);
    ws.WihT0[packW(k, j, G4 >> 4)] = (bf16_t)pWih0[(size_t)j * D2 + k];
  }
  for (size_t i = t0; i < (size_t)D * G4; i += stride) {
    int k = (int)(i / G4), j = (int)(i % G4);
    ws.WhhT0[packW(k, j, G4 >> 4)] = (bf16_t)pWhh0[(size_t)j * D + k];
    ws.WihT1[packW(k, j, G4 >> 4)] = (bf16_t)pWih1[(size_t)j * D + k];
    ws.WhhT1[packW(k, j, G4 >> 4)] = (bf16_t)pWhh1[(size_t)j * D + k];
  }
  for (size_t i = t0; i < (size_t)D * D; i += stride) {
    int k = (int)(i / D), j = (int)(i % D);
    ws.Wqu[packW(k, j, D >> 4)] = (bf16_t)pWqu[i];
    ws.Wqc[packW(k, j, D >> 4)] = (bf16_t)pWqc[i];
  }
  for (size_t i = t0; i < (size_t)D2 * D; i += stride) {
    int k = (int)(i / D), j = (int)(i % D);
    ws.WoutT[packW(k, j, D >> 4)] = (bf16_t)pWout[(size_t)j * D2 + k];
  }
}

// keys/vals/chunks f32 -> bf16 (layout unchanged: [rows][B][D])
__global__ void prep_kv_kernel(bf16_t* __restrict__ kbf, bf16_t* __restrict__ vbf,
                               bf16_t* __restrict__ cbf,
                               const float* __restrict__ keys,
                               const float* __restrict__ vals,
                               const float* __restrict__ chunks) {
  const size_t stride = (size_t)gridDim.x * blockDim.x;
  const size_t t0 = (size_t)blockIdx.x * blockDim.x + threadIdx.x;
  for (size_t i = t0; i < (size_t)NU * B * D; i += stride) {
    kbf[i] = (bf16_t)keys[i];
    vbf[i] = (bf16_t)vals[i];
  }
  for (size_t i = t0; i < (size_t)NE * B * D; i += stride)
    cbf[i] = (bf16_t)chunks[i];
}

// ---------------- persistent decoder ----------------
__global__ __launch_bounds__(TPB) void decoder_kernel(
    WS ws, const float* __restrict__ pos_embs, const float* __restrict__ units,
    const float* __restrict__ chunks, const bf16_t* __restrict__ kbf,
    const bf16_t* __restrict__ vbf, const bf16_t* __restrict__ cbf,
    const unsigned char* __restrict__ umask, const unsigned char* __restrict__ cmask,
    const float* __restrict__ bih0, const float* __restrict__ bhh0,
    const float* __restrict__ bih1, const float* __restrict__ bhh1,
    float* __restrict__ outs, float* __restrict__ aligns) {
  __shared__ float s_align[NU];  // 8 KB
#if HAVE_ASYNC_LDS
  __shared__ bf16_t s_vtile[2][8 * D];  // 2 x 8 KB double buffer for vals tiles
#endif

  const int lane = threadIdx.x & 31;
  const int wid  = blockIdx.x * (TPB / 32) + (threadIdx.x >> 5);
  const int gtid = blockIdx.x * TPB + threadIdx.x;
  const bool usebf = (kbf != nullptr);
  unsigned gen = 0;

  for (int t = 0; t < T_STEPS; ++t) {
    // ---- P1: LSTM0 gates  g0 = [emb_t|feed] @ WihT0 + h0 @ WhhT0 + biases
    {
      int btile = wid >> 7, ntile = wid & 127;  // 4 x 128 = 512 wave jobs
      int b0 = btile * 16, j0 = ntile * 16;
      float bias = bih0[j0 + (lane & 15)] + bhh0[j0 + (lane & 15)];
      v8f acc;
#pragma unroll
      for (int r = 0; r < 8; ++r) acc[r] = bias;
      const bf16_t* embt = ws.embT + (size_t)t * B * D;
      for (int kt = 0; kt < 32; ++kt) {  // K = 1024 (x = emb|feed)
        int k0 = kt * 32;
        const bf16_t* asrc = (k0 < D) ? (embt + (size_t)b0 * D + k0)
                                      : (ws.feedbf + (size_t)b0 * D + (k0 - D));
        v16bf a  = load_a_frag(asrc, D, lane);
        v16bf bm = load_b_frag(ws.WihT0 + packW(k0, j0, G4 >> 4), lane);
        acc = wmma_bf16(a, bm, acc);
      }
      for (int kt = 0; kt < 16; ++kt) {  // K = 512 (h0 recurrent)
        int k0 = kt * 32;
        v16bf a  = load_a_frag(ws.h0bf + (size_t)b0 * D + k0, D, lane);
        v16bf bm = load_b_frag(ws.WhhT0 + packW(k0, j0, G4 >> 4), lane);
        acc = wmma_bf16(a, bm, acc);
      }
      store_c_frag(ws.g0 + (size_t)b0 * G4 + j0, G4, lane, acc);
    }
    grid_sync(ws.cnt, gen);

    // ---- P2: pointwise LSTM0
    for (int idx = gtid; idx < B * D; idx += NTHREADS) {
      int b = idx >> 9, d = idx & (D - 1);
      const float* g = ws.g0 + (size_t)b * G4;
      float ig = sigmoidf_(g[d]);
      float fg = sigmoidf_(g[D + d]);
      float gg = tanhf(g[2 * D + d]);
      float og = sigmoidf_(g[3 * D + d]);
      float c = fg * ws.c0[idx] + ig * gg;
      ws.c0[idx] = c;
      ws.h0bf[idx] = (bf16_t)(og * tanhf(c));
    }
    grid_sync(ws.cnt, gen);

    // ---- P3: LSTM1 gates
    {
      int btile = wid >> 7, ntile = wid & 127;
      int b0 = btile * 16, j0 = ntile * 16;
      float bias = bih1[j0 + (lane & 15)] + bhh1[j0 + (lane & 15)];
      v8f acc;
#pragma unroll
      for (int r = 0; r < 8; ++r) acc[r] = bias;
      for (int kt = 0; kt < 16; ++kt) {
        int k0 = kt * 32;
        v16bf a  = load_a_frag(ws.h0bf + (size_t)b0 * D + k0, D, lane);
        v16bf bm = load_b_frag(ws.WihT1 + packW(k0, j0, G4 >> 4), lane);
        acc = wmma_bf16(a, bm, acc);
      }
      for (int kt = 0; kt < 16; ++kt) {
        int k0 = kt * 32;
        v16bf a  = load_a_frag(ws.h1bf + (size_t)b0 * D + k0, D, lane);
        v16bf bm = load_b_frag(ws.WhhT1 + packW(k0, j0, G4 >> 4), lane);
        acc = wmma_bf16(a, bm, acc);
      }
      store_c_frag(ws.g1 + (size_t)b0 * G4 + j0, G4, lane, acc);
    }
    grid_sync(ws.cnt, gen);

    // ---- P4: pointwise LSTM1 -> q
    for (int idx = gtid; idx < B * D; idx += NTHREADS) {
      int b = idx >> 9, d = idx & (D - 1);
      const float* g = ws.g1 + (size_t)b * G4;
      float ig = sigmoidf_(g[d]);
      float fg = sigmoidf_(g[D + d]);
      float gg = tanhf(g[2 * D + d]);
      float og = sigmoidf_(g[3 * D + d]);
      float c = fg * ws.c1[idx] + ig * gg;
      ws.c1[idx] = c;
      bf16_t hb = (bf16_t)(og * tanhf(c));
      ws.h1bf[idx] = hb;
      ws.ctxqbf[(size_t)b * D2 + D + d] = hb;
    }
    grid_sync(ws.cnt, gen);

    // ---- P5: q projections
    if (wid < 256) {
      const bf16_t* Wm = (wid < 128) ? ws.Wqu : ws.Wqc;
      float* Om        = (wid < 128) ? ws.qWu : ws.qWc;
      int lw = wid & 127;
      int b0 = (lw >> 5) * 16, j0 = (lw & 31) * 16;
      v8f acc;
#pragma unroll
      for (int r = 0; r < 8; ++r) acc[r] = 0.0f;
      for (int kt = 0; kt < 16; ++kt) {
        int k0 = kt * 32;
        v16bf a  = load_a_frag(ws.h1bf + (size_t)b0 * D + k0, D, lane);
        v16bf bm = load_b_frag(Wm + packW(k0, j0, D >> 4), lane);
        acc = wmma_bf16(a, bm, acc);
      }
      store_c_frag(Om + (size_t)b0 * D + j0, D, lane, acc);
    }
    grid_sync(ws.cnt, gen);

    // ---- P6: attention scores (bandwidth-bound key stream, bf16 if possible)
    {
      const int total = B * NU + B * NE;
      for (int r = wid; r < total; r += NWAVES) {
        if (r < B * NU) {
          int b = r >> 11, u = r & (NU - 1);
          const float* qp = ws.qWu + (size_t)b * D;
          float s = usebf
              ? rowdot_bf16(qp, kbf + (size_t)u * B * D + (size_t)b * D, lane)
              : rowdot_f32(qp, pos_embs + (size_t)u * B * D + (size_t)b * D, lane);
          if (lane == 0)
            ws.su[(size_t)b * NU + u] = umask[(size_t)b * NU + u] ? s : NEGINF;
        } else {
          int r2 = r - B * NU;
          int b = r2 >> 6, e = r2 & (NE - 1);
          const float* qp = ws.qWc + (size_t)b * D;
          float s = usebf
              ? rowdot_bf16(qp, cbf + (size_t)e * B * D + (size_t)b * D, lane)
              : rowdot_f32(qp, chunks + (size_t)e * B * D + (size_t)b * D, lane);
          if (lane == 0)
            ws.sc[(size_t)b * NE + e] = cmask[(size_t)b * NE + e] ? s : NEGINF;
        }
      }
    }
    grid_sync(ws.cnt, gen);

    // ---- P7a: within-entity softmax (au) and entity softmax (ac)
    for (int r = wid; r < B * NE; r += NWAVES) {
      int b = r >> 6, e = r & (NE - 1);
      size_t base = (size_t)b * NU + e * ES;
      float v = ws.su[base + lane];
      float m = wave_max(v);
      float p = expf(v - m);
      float s = wave_sum(p);
      ws.au[base + lane] = p / s;
    }
    if (wid < B) {
      int b = wid;
      float v0 = ws.sc[(size_t)b * NE + lane];
      float v1 = ws.sc[(size_t)b * NE + 32 + lane];
      float m = wave_max(fmaxf(v0, v1));
      float p0 = expf(v0 - m), p1 = expf(v1 - m);
      float s = wave_sum(p0 + p1);
      ws.ac[(size_t)b * NE + lane] = p0 / s;
      ws.ac[(size_t)b * NE + 32 + lane] = p1 / s;
    }
    grid_sync(ws.cnt, gen);

    // ---- P7b: align = ac * au  -> straight to output buffer
    {
      float* at = aligns + (size_t)t * B * NU;
      for (int idx = gtid; idx < B * NU; idx += NTHREADS) {
        int b = idx >> 11, u = idx & (NU - 1);
        at[idx] = ws.ac[(size_t)b * NE + (u >> 5)] * ws.au[idx];
      }
    }
    grid_sync(ws.cnt, gen);

    // ---- P8: ctx[b] = sum_u align[b,u] * vals[b,u,:]  (one WG per batch)
    {
      int b = blockIdx.x;
      const float* at = aligns + (size_t)t * B * NU + (size_t)b * NU;
      for (int i = threadIdx.x; i < NU; i += TPB) s_align[i] = at[i];
      __syncthreads();

      if (usebf) {
#if HAVE_ASYNC_LDS
        // double-buffered async-to-LDS tiles of 8 rows (8 KB each)
        const bf16_t* vb = vbf + (size_t)b * D;
        const int o  = threadIdx.x * 32;  // byte offset within 8 KB tile
        const int tr = o >> 10, to = o & 1023;
        {
          const char* g = (const char*)(vb + (size_t)tr * B * D) + to;
          char* l = (char*)&s_vtile[0][0] + o;
          async_copy_b128(g, l);
          async_copy_b128(g + 16, l + 16);
        }
        float a0 = 0.0f, a1 = 0.0f;
        for (int ut = 0; ut < NU / 8; ++ut) {
          int cur = ut & 1;
          wait_async0();
          __syncthreads();
          if (ut + 1 < NU / 8) {
            const char* g =
                (const char*)(vb + (size_t)((ut + 1) * 8 + tr) * B * D) + to;
            char* l = (char*)&s_vtile[cur ^ 1][0] + o;
            async_copy_b128(g, l);
            async_copy_b128(g + 16, l + 16);
          }
          const unsigned* lt = (const unsigned*)&s_vtile[cur][0];
#pragma unroll
          for (int r = 0; r < 8; ++r) {
            float w = s_align[ut * 8 + r];
            union { unsigned u32; bf16_t h[2]; } cv;
            cv.u32 = lt[r * (D / 2) + threadIdx.x];
            a0 = fmaf(w, (float)cv.h[0], a0);
            a1 = fmaf(w, (float)cv.h[1], a1);
          }
        }
        ws.ctxqbf[(size_t)b * D2 + 2 * threadIdx.x]     = (bf16_t)a0;
        ws.ctxqbf[(size_t)b * D2 + 2 * threadIdx.x + 1] = (bf16_t)a1;
#else
        // direct bf16 stream: packed 2 x bf16 per thread (b32, coalesced)
        float a0 = 0.0f, a1 = 0.0f;
        for (int u = 0; u < NU; ++u) {
          const unsigned* row =
              (const unsigned*)(vbf + (size_t)u * B * D + (size_t)b * D);
          __builtin_prefetch(row + (size_t)(B * D) / 2, 0, 1);
          union { unsigned u32; bf16_t h[2]; } cv;
          cv.u32 = row[threadIdx.x];
          float w = s_align[u];
          a0 = fmaf(w, (float)cv.h[0], a0);
          a1 = fmaf(w, (float)cv.h[1], a1);
        }
        ws.ctxqbf[(size_t)b * D2 + 2 * threadIdx.x]     = (bf16_t)a0;
        ws.ctxqbf[(size_t)b * D2 + 2 * threadIdx.x + 1] = (bf16_t)a1;
#endif
      } else {
        // f32 fallback stream
        int d0 = threadIdx.x, d1 = threadIdx.x + TPB;
        float a0 = 0.0f, a1 = 0.0f;
        for (int u = 0; u < NU; ++u) {
          const float* row = units + (size_t)u * B * D + (size_t)b * D;
          __builtin_prefetch(row + (size_t)B * D, 0, 1);
          float w = s_align[u];
          a0 = fmaf(w, row[d0], a0);
          a1 = fmaf(w, row[d1], a1);
        }
        ws.ctxqbf[(size_t)b * D2 + d0] = (bf16_t)a0;
        ws.ctxqbf[(size_t)b * D2 + d1] = (bf16_t)a1;
      }
    }
    grid_sync(ws.cnt, gen);

    // ---- P9: out = tanh(concat(ctx,q) @ WoutT) -> dec_outs[t+1], feed
    if (wid < 128) {
      int b0 = (wid >> 5) * 16, j0 = (wid & 31) * 16;
      v8f acc;
#pragma unroll
      for (int r = 0; r < 8; ++r) acc[r] = 0.0f;
      for (int kt = 0; kt < 32; ++kt) {
        int k0 = kt * 32;
        v16bf a  = load_a_frag(ws.ctxqbf + (size_t)b0 * D2 + k0, D2, lane);
        v16bf bm = load_b_frag(ws.WoutT + packW(k0, j0, D >> 4), lane);
        acc = wmma_bf16(a, bm, acc);
      }
      int col = lane & 15, rb = (lane < 16) ? 0 : 8;
#pragma unroll
      for (int r = 0; r < 8; ++r) {
        float o = tanhf(acc[r]);
        int b = b0 + rb + r, j = j0 + col;
        outs[(size_t)(t + 1) * B * D + (size_t)b * D + j] = o;
        ws.feedbf[(size_t)b * D + j] = (bf16_t)o;
      }
    }
    grid_sync(ws.cnt, gen);
  }
}

// ---------------- host launcher ----------------
extern "C" void kernel_launch(void* const* d_in, const int* in_sizes, int n_in,
                              void* d_out, int out_size, void* d_ws,
                              size_t ws_size, hipStream_t stream) {
  (void)in_sizes; (void)n_in; (void)out_size;

  const int*   tgt       = (const int*)d_in[0];
  const float* chunks    = (const float*)d_in[1];
  const float* units     = (const float*)d_in[2];
  const float* pos_embs  = (const float*)d_in[3];
  const unsigned char* umask = (const unsigned char*)d_in[4];
  const unsigned char* cmask = (const unsigned char*)d_in[5];
  const float* enc       = (const float*)d_in[6];
  const float* emb_table = (const float*)d_in[7];
  const float* sif       = (const float*)d_in[8];
  const float* Wih0 = (const float*)d_in[9];
  const float* Whh0 = (const float*)d_in[10];
  const float* bih0 = (const float*)d_in[11];
  const float* bhh0 = (const float*)d_in[12];
  const float* Wih1 = (const float*)d_in[13];
  const float* Whh1 = (const float*)d_in[14];
  const float* bih1 = (const float*)d_in[15];
  const float* bhh1 = (const float*)d_in[16];
  const float* Wqu  = (const float*)d_in[17];
  const float* Wqc  = (const float*)d_in[18];
  const float* Wout = (const float*)d_in[19];

  float* outs   = (float*)d_out;                          // [T+1][B][D]
  float* aligns = outs + (size_t)(T_STEPS + 1) * B * D;   // [T][B][NU]

  char* p = (char*)d_ws;
  auto carve = [&](size_t bytes) -> char* {
    char* r = p;
    p += (bytes + 255) & ~(size_t)255;
    return r;
  };
  WS ws;
  ws.embT   = (bf16_t*)carve((size_t)T_STEPS * B * D * 2);
  ws.WihT0  = (bf16_t*)carve((size_t)D2 * G4 * 2);
  ws.WhhT0  = (bf16_t*)carve((size_t)D * G4 * 2);
  ws.WihT1  = (bf16_t*)carve((size_t)D * G4 * 2);
  ws.WhhT1  = (bf16_t*)carve((size_t)D * G4 * 2);
  ws.Wqu    = (bf16_t*)carve((size_t)D * D * 2);
  ws.Wqc    = (bf16_t*)carve((size_t)D * D * 2);
  ws.WoutT  = (bf16_t*)carve((size_t)D2 * D * 2);
  ws.h0bf   = (bf16_t*)carve((size_t)B * D * 2);
  ws.h1bf   = (bf16_t*)carve((size_t)B * D * 2);
  ws.feedbf = (bf16_t*)carve((size_t)B * D * 2);
  ws.ctxqbf = (bf16_t*)carve((size_t)B * D2 * 2);
  ws.g0  = (float*)carve((size_t)B * G4 * 4);
  ws.g1  = (float*)carve((size_t)B * G4 * 4);
  ws.c0  = (float*)carve((size_t)B * D * 4);
  ws.c1  = (float*)carve((size_t)B * D * 4);
  ws.qWu = (float*)carve((size_t)B * D * 4);
  ws.qWc = (float*)carve((size_t)B * D * 4);
  ws.su  = (float*)carve((size_t)B * NU * 4);
  ws.sc  = (float*)carve((size_t)B * NE * 4);
  ws.au  = (float*)carve((size_t)B * NU * 4);
  ws.ac  = (float*)carve((size_t)B * NE * 4);
  ws.cnt = (unsigned*)carve(256);

  // optional bf16 copies of the streamed tensors (the roofline-dominant data)
  bf16_t *kbf = nullptr, *vbf = nullptr, *cbf = nullptr;
  {
    bf16_t* tk = (bf16_t*)carve((size_t)NU * B * D * 2);
    bf16_t* tv = (bf16_t*)carve((size_t)NU * B * D * 2);
    bf16_t* tc = (bf16_t*)carve((size_t)NE * B * D * 2);
    if ((size_t)(p - (char*)d_ws) <= ws_size) {
      kbf = tk; vbf = tv; cbf = tc;
    }
  }

  prep_state_kernel<<<(B * D + TPB - 1) / TPB, TPB, 0, stream>>>(ws, enc, sif, outs);
  prep_emb_kernel<<<512, TPB, 0, stream>>>(ws, tgt, emb_table);
  prep_weights_kernel<<<1024, TPB, 0, stream>>>(ws, Wih0, Whh0, Wih1, Whh1, Wqu,
                                                Wqc, Wout);
  if (kbf)
    prep_kv_kernel<<<2048, TPB, 0, stream>>>(kbf, vbf, cbf, pos_embs, units, chunks);
  decoder_kernel<<<NWG, TPB, 0, stream>>>(ws, pos_embs, units, chunks, kbf, vbf,
                                          cbf, umask, cmask, bih0, bhh0, bih1,
                                          bhh1, outs, aligns);
}
